// CausalSelfAttention_67946382623081
// MI455X (gfx1250) — compile-verified, matrix operand order checked by reading, not run on hardware
//
#include <hip/hip_runtime.h>

// Problem constants (match reference): B=4, T=2048, C=1024, H=16, D=64
#define BB 4
#define TT 2048
#define CC 1024
#define HH 16
#define DD 64

typedef __attribute__((ext_vector_type(16))) __bf16 v16bf;
typedef __attribute__((ext_vector_type(8)))  float  v8f;

union FragU { v16bf v; uint4 q[2]; };

__device__ __forceinline__ unsigned short f2bf(float f) {
  unsigned int u = __builtin_bit_cast(unsigned int, f);
  u += 0x7FFFu + ((u >> 16) & 1u);          // round-to-nearest-even
  return (unsigned short)(u >> 16);
}

// Load a 16-element bf16 fragment for WMMA 16x16x32 (A or B operand).
// Per ISA layout: lane-half h covers K = kbase+8h..+7 and kbase+16+8h..+7.
__device__ __forceinline__ v16bf load_frag(const unsigned short* base,
                                           int ld, int row, int kbase, int half) {
  const unsigned short* p = base + (size_t)row * ld + kbase + 8 * half;
  FragU u;
  u.q[0] = *reinterpret_cast<const uint4*>(p);        // 16B (global or ds b128)
  u.q[1] = *reinterpret_cast<const uint4*>(p + 16);   // 16B
  return u.v;
}

__device__ __forceinline__ v8f wmma_bf16(v16bf a, v16bf b, v8f c) {
  return __builtin_amdgcn_wmma_f32_16x16x32_bf16(false, a, false, b, (short)0, c,
                                                 false, false);
}

// Async global->LDS 16B transfer (gfx1250, ASYNCcnt-tracked).
// ISA 15.18.3 op 98: VDST = per-lane LDS byte address, VADDR = 64-bit address.
__device__ __forceinline__ void async_cp16(const unsigned short* g, unsigned short* l) {
  unsigned lds_off = (unsigned)(size_t)l;
  unsigned long long ga = (unsigned long long)(size_t)g;
  asm volatile("global_load_async_to_lds_b128 %0, %1, off"
               :: "v"(lds_off), "v"(ga) : "memory");
}
__device__ __forceinline__ void wait_async0() {
  asm volatile("s_wait_asynccnt 0x0" ::: "memory");
}

// ---------------------------------------------------------------- converters
__global__ void cvt_f32_bf16(const float* __restrict__ in,
                             unsigned short* __restrict__ out, int n8) {
  int i = blockIdx.x * blockDim.x + threadIdx.x;
  if (i >= n8) return;
  const float4* p = reinterpret_cast<const float4*>(in) + 2 * (size_t)i;
  float4 a = p[0], b = p[1];
  uint4 r;
  r.x = (unsigned)f2bf(a.x) | ((unsigned)f2bf(a.y) << 16);
  r.y = (unsigned)f2bf(a.z) | ((unsigned)f2bf(a.w) << 16);
  r.z = (unsigned)f2bf(b.x) | ((unsigned)f2bf(b.y) << 16);
  r.w = (unsigned)f2bf(b.z) | ((unsigned)f2bf(b.w) << 16);
  reinterpret_cast<uint4*>(out)[i] = r;
}

// w [K x N] fp32 -> wt [N x K] bf16  (B^T storage so B-frags are contiguous)
__global__ void transpose_cvt(const float* __restrict__ w,
                              unsigned short* __restrict__ wt, int K, int N) {
  int i = blockIdx.x * blockDim.x + threadIdx.x;
  if (i >= K * N) return;
  int n = i / K, c = i - n * K;
  wt[i] = f2bf(w[(size_t)c * N + n]);
}

// ---------------------------------------------------------------- WMMA GEMM
// 128-thread block (4 waves) computes a 128(M) x 128(N) tile; each wave 64x64.
// A/B K=32 slices are staged in double-buffered LDS via async global->LDS DMA,
// overlapping the next slice's DMA with the current slice's 16 WMMAs per wave.
// MODE 0: QKV epilogue -> scatter bf16 into q[B,H,T,D], k[B,H,T,D], vT[B,H,D,T]
// MODE 1: proj epilogue -> fp32 out[M,N] + bias
template <int MODE>
__global__ void wmma_gemm(const unsigned short* __restrict__ A,   // [M x K] bf16
                          const unsigned short* __restrict__ Bt,  // [N x K] bf16
                          const float* __restrict__ bias,         // [N]
                          float* __restrict__ outF,
                          unsigned short* __restrict__ qb,
                          unsigned short* __restrict__ kb,
                          unsigned short* __restrict__ vtb,
                          int M, int N, int K) {
  __shared__ __align__(16) unsigned short shA[2][128 * 32];   // 2 x 8 KiB
  __shared__ __align__(16) unsigned short shB[2][128 * 32];   // 2 x 8 KiB

  const int tid  = threadIdx.x;
  const int lane = tid & 31;
  const int lrow = lane & 15;
  const int half = lane >> 4;
  const int w    = tid >> 5;          // wave id 0..3
  const int wm   = (w >> 1) * 64;     // wave origin in block tile (2x2)
  const int wn   = (w & 1) * 64;

  const int tilesN = N >> 7;
  const int tm = blockIdx.x / tilesN;
  const int tn = blockIdx.x - tm * tilesN;
  const int m0 = tm * 128, n0 = tn * 128;

  // --- async stage of one K=32 slice: 512 16B-chunks per tile, 128 threads
  auto stage = [&](int buf, int kk) {
#pragma unroll
    for (int c = 0; c < 4; ++c) {
      int ch  = tid + c * 128;         // 0..511
      int row = ch >> 2;               // 0..127
      int ko  = (ch & 3) * 8;          // 0,8,16,24 (bf16 elems)
      async_cp16(A  + (size_t)(m0 + row) * K + kk + ko, &shA[buf][row * 32 + ko]);
      async_cp16(Bt + (size_t)(n0 + row) * K + kk + ko, &shB[buf][row * 32 + ko]);
    }
  };

  v8f acc[4][4];
#pragma unroll
  for (int i = 0; i < 4; ++i)
#pragma unroll
    for (int j = 0; j < 4; ++j) acc[i][j] = (v8f)(0.0f);

  const int S = K >> 5;               // K=1024 -> 32 slices
  stage(0, 0);
  wait_async0();
  __syncthreads();

  for (int s = 0; s < S; ++s) {
    const int buf = s & 1;
    if (s + 1 < S) stage(buf ^ 1, (s + 1) << 5);   // DMA next slice (overlaps)

    v16bf a[4];
#pragma unroll
    for (int i = 0; i < 4; ++i)
      a[i] = load_frag(shA[buf], 32, wm + i * 16 + lrow, 0, half);
#pragma unroll
    for (int j = 0; j < 4; ++j) {
      v16bf b = load_frag(shB[buf], 32, wn + j * 16 + lrow, 0, half);
#pragma unroll
      for (int i = 0; i < 4; ++i) acc[i][j] = wmma_bf16(a[i], b, acc[i][j]);
    }

    wait_async0();                    // next slice landed in LDS
    __syncthreads();                  // everyone done reading buf before reuse
  }

#pragma unroll
  for (int i = 0; i < 4; ++i)
#pragma unroll
    for (int j = 0; j < 4; ++j) {
#pragma unroll
      for (int r = 0; r < 8; ++r) {
        int gm = m0 + wm + 16 * i + r + 8 * half;  // C/D layout: M = r + 8*(lane/16)
        int gn = n0 + wn + 16 * j + lrow;          // N = lane%16
        float val = acc[i][j][r] + bias[gn];
        if (MODE == 1) {
          outF[(size_t)gm * N + gn] = val;
        } else {
          int b    = gm >> 11;                     // / T
          int t    = gm & (TT - 1);
          int part = gn >> 10;                     // / C  (0:q 1:k 2:v)
          int c    = gn & (CC - 1);
          int h    = c >> 6;                       // / D
          int d    = c & (DD - 1);
          unsigned short bv = f2bf(val);
          size_t bh = (size_t)(b * HH + h);
          if (part == 0)      qb [(bh * TT + t) * DD + d] = bv;
          else if (part == 1) kb [(bh * TT + t) * DD + d] = bv;
          else                vtb[(bh * DD + d) * TT + t] = bv;   // V transposed
        }
      }
    }
}

// ---------------------------------------------------------------- attention
// One wave per (b,h, 16-row q-tile). Flash-style single pass, k-chunks of 32.
__global__ void attn_kernel(const unsigned short* __restrict__ q,
                            const unsigned short* __restrict__ k,
                            const unsigned short* __restrict__ vT,
                            unsigned short* __restrict__ y) {
  const int qtiles = TT / 16;
  const int qt = blockIdx.x % qtiles;
  const int bh = blockIdx.x / qtiles;
  const int b = bh / HH, h = bh - b * HH;

  const unsigned short* Q  = q  + (size_t)bh * TT * DD;
  const unsigned short* Kp = k  + (size_t)bh * TT * DD;
  const unsigned short* Vt = vT + (size_t)bh * DD * TT;

  const int lane = threadIdx.x & 31;
  const int lrow = lane & 15;
  const int half = lane >> 4;
  const int q0 = qt * 16;

  // Q A-fragments for K-dim = D = 64 (two 32-chunks); fixed for whole pass
  v16bf qa0 = load_frag(Q, DD, q0 + lrow, 0, half);
  v16bf qa1 = load_frag(Q, DD, q0 + lrow, 32, half);

  v8f o[4];
#pragma unroll
  for (int j = 0; j < 4; ++j) o[j] = (v8f)(0.0f);
  float mstate[8], lstate[8];
#pragma unroll
  for (int r = 0; r < 8; ++r) { mstate[r] = -1e30f; lstate[r] = 0.0f; }

  __shared__ __align__(16) unsigned short pl[16 * 32];  // P tile staging

  const int kend = q0 + 15;
  for (int kc = 0; kc <= kend; kc += 32) {
    if (kc + 32 <= kend) {            // prefetch next chunk (global_prefetch_b8)
      __builtin_prefetch(Kp + (size_t)(kc + 32 + lrow + 16 * half) * DD, 0, 0);
      __builtin_prefetch(Vt + (size_t)(lrow + 16 * half) * TT + kc + 32, 0, 0);
    }

    // ---- S = Q @ K^T  (two 16x16 n-tiles, K-dim 64)
    v8f s[2];
#pragma unroll
    for (int t = 0; t < 2; ++t) {
      s[t] = (v8f)(0.0f);
      v16bf kb0 = load_frag(Kp, DD, kc + t * 16 + lrow, 0, half);
      v16bf kb1 = load_frag(Kp, DD, kc + t * 16 + lrow, 32, half);
      s[t] = wmma_bf16(qa0, kb0, s[t]);
      s[t] = wmma_bf16(qa1, kb1, s[t]);
    }

    // ---- scale + causal mask
    float sv[2][8];
#pragma unroll
    for (int t = 0; t < 2; ++t)
#pragma unroll
      for (int r = 0; r < 8; ++r) {
        float v = s[t][r] * 0.125f;                 // 1/sqrt(64)
        int row = q0 + r + 8 * half;
        int col = kc + 16 * t + lrow;
        if (col > row) v = -1e30f;
        sv[t][r] = v;
      }

    // ---- online softmax: row max / sum over the 16-lane group
    float cm[8];
#pragma unroll
    for (int r = 0; r < 8; ++r) {
      float v = fmaxf(sv[0][r], sv[1][r]);
      v = fmaxf(v, __shfl_xor(v, 1));
      v = fmaxf(v, __shfl_xor(v, 2));
      v = fmaxf(v, __shfl_xor(v, 4));
      v = fmaxf(v, __shfl_xor(v, 8));
      cm[r] = v;
    }
    float alpha[8];
#pragma unroll
    for (int r = 0; r < 8; ++r) {
      float mn = fmaxf(mstate[r], cm[r]);
      alpha[r] = __expf(mstate[r] - mn);
      mstate[r] = mn;
    }
#pragma unroll
    for (int t = 0; t < 2; ++t)
#pragma unroll
      for (int r = 0; r < 8; ++r) sv[t][r] = __expf(sv[t][r] - mstate[r]);
#pragma unroll
    for (int r = 0; r < 8; ++r) {
      float v = sv[0][r] + sv[1][r];
      v += __shfl_xor(v, 1);
      v += __shfl_xor(v, 2);
      v += __shfl_xor(v, 4);
      v += __shfl_xor(v, 8);
      lstate[r] = lstate[r] * alpha[r] + v;
    }
#pragma unroll
    for (int j = 0; j < 4; ++j)
#pragma unroll
      for (int r = 0; r < 8; ++r) o[j][r] *= alpha[r];

    // ---- stage P (16x32 bf16) through LDS to reshape C-layout -> A-frag
#pragma unroll
    for (int t = 0; t < 2; ++t)
#pragma unroll
      for (int r = 0; r < 8; ++r)
        pl[(r + 8 * half) * 32 + 16 * t + lrow] = f2bf(sv[t][r]);
    __syncthreads();
    v16bf pa = load_frag(pl, 32, lrow, 0, half);

    // ---- O += P @ V  (4 d-tiles, K-dim 32; Vt is [D,T] so B-frags contiguous)
#pragma unroll
    for (int j = 0; j < 4; ++j) {
      v16bf vb = load_frag(Vt, TT, j * 16 + lrow, kc, half);
      o[j] = wmma_bf16(pa, vb, o[j]);
    }
    __syncthreads();
  }

  // ---- epilogue: O / l -> y[B*T, C] bf16
#pragma unroll
  for (int j = 0; j < 4; ++j)
#pragma unroll
    for (int r = 0; r < 8; ++r) {
      float val = o[j][r] / lstate[r];
      int t = q0 + r + 8 * half;
      y[((size_t)(b * TT + t)) * CC + h * DD + j * 16 + lrow] = f2bf(val);
    }
}

// ---------------------------------------------------------------- launch
extern "C" void kernel_launch(void* const* d_in, const int* in_sizes, int n_in,
                              void* d_out, int out_size, void* d_ws, size_t ws_size,
                              hipStream_t stream) {
  const float* x      = (const float*)d_in[0];
  const float* w_qkv  = (const float*)d_in[1];
  const float* b_qkv  = (const float*)d_in[2];
  const float* w_proj = (const float*)d_in[3];
  const float* b_proj = (const float*)d_in[4];
  float* out = (float*)d_out;

  const size_t M = (size_t)BB * TT;          // 8192
  char* ws = (char*)d_ws;
  unsigned short* xb     = (unsigned short*)ws;  ws += M * CC * 2;              // 16 MiB
  unsigned short* wqkvT  = (unsigned short*)ws;  ws += (size_t)3 * CC * CC * 2; //  6 MiB
  unsigned short* wprojT = (unsigned short*)ws;  ws += (size_t)CC * CC * 2;     //  2 MiB
  unsigned short* qb     = (unsigned short*)ws;  ws += M * CC * 2;              // 16 MiB
  unsigned short* kb     = (unsigned short*)ws;  ws += M * CC * 2;              // 16 MiB
  unsigned short* vtb    = (unsigned short*)ws;  ws += M * CC * 2;              // 16 MiB
  unsigned short* yb     = (unsigned short*)ws;                                 // 16 MiB

  // 1) fp32 -> bf16 conversions (x) and weight transposes ([K,N]->[N,K])
  {
    int n8 = (int)(M * CC / 8);
    cvt_f32_bf16<<<(n8 + 255) / 256, 256, 0, stream>>>(x, xb, n8);
  }
  {
    int n = CC * 3 * CC;
    transpose_cvt<<<(n + 255) / 256, 256, 0, stream>>>(w_qkv, wqkvT, CC, 3 * CC);
  }
  {
    int n = CC * CC;
    transpose_cvt<<<(n + 255) / 256, 256, 0, stream>>>(w_proj, wprojT, CC, CC);
  }

  // 2) QKV GEMM: [8192,1024] x [1024,3072] -> scatter q/k/vT (bf16)
  {
    int blocks = (int)(M / 128) * ((3 * CC) / 128);   // 64 * 24 = 1536
    wmma_gemm<0><<<blocks, 128, 0, stream>>>(xb, wqkvT, b_qkv, nullptr,
                                             qb, kb, vtb, (int)M, 3 * CC, CC);
  }

  // 3) Flash attention: one wave per (b,h,q-tile)
  {
    int blocks = BB * HH * (TT / 16);                 // 8192
    attn_kernel<<<blocks, 32, 0, stream>>>(qb, kb, vtb, yb);
  }

  // 4) Output projection: [8192,1024] x [1024,1024] + bias -> fp32 out
  {
    int blocks = (int)(M / 128) * (CC / 128);         // 64 * 8 = 512
    wmma_gemm<1><<<blocks, 128, 0, stream>>>(yb, wprojT, b_proj, out,
                                             nullptr, nullptr, nullptr,
                                             (int)M, CC, CC);
  }
}
